// GraphTMixer_73203422593197
// MI455X (gfx1250) — matrix operand order repaired and checked by reading, not compiled, not run on hardware
//
#include <hip/hip_runtime.h>
#include <hip/hip_bf16.h>
#include <math.h>

// ---------------------------------------------------------------------------
// GraphTMixer for MI455X (gfx1250, wave32, WMMA bf16 16x16x32, f32 accum)
//
// Pipeline (B=8, T=96, N=1024, C=64, H=64, BT=768, BN=8192):
//  K-casts : adj, x, w1, w2 -> bf16
//  K-nodeW : A[n,o] = sum_i w_in[i]*node_w[n,i,o]; Bv likewise with b_in
//  K-rsum  : r[n] = sum_m adj[n,m]
//  K-gemmS : S[n,bt] = sum_m adj[n,m]*x[bt,m]   (WMMA bf16, 1024x768x1024)
//  K-gcngn : h[bt,c,n] = s*A + r*Bv + x*w_in + b_in, GroupNorm(4,C) over
//            (c-in-group, n); write hn[b,n,t,c] as bf16 (temporal layout)
//  K-mixer : per (b,n): Z = relu(Ht @ W1^T + b1) ; Tt = Z @ W2^T + b2 (WMMA),
//            y[t] = b_out + sum_c silu(hn[t,c] + Tt[c,t]) * w_out[c]
// ---------------------------------------------------------------------------

typedef __bf16 bf16;
typedef __attribute__((ext_vector_type(16))) __bf16 v16bf;
typedef __attribute__((ext_vector_type(8)))  float  v8f;

#define BB 8
#define TT 96
#define NN 1024
#define CC 64
#define HH 64
#define BT 768
#define BN 8192

// ---- WMMA fragment helpers (layouts per CDNA5 ISA 7.12.2, wave32) ---------

// A-fragment 16x32 bf16 from row-major matrix, leading dim ld (elements).
// lane L: row = L&15; lane-half h=L>>4: elems 0..7 = K h*8..h*8+7,
//                                       elems 8..15 = K 16+h*8 .. 16+h*8+7
__device__ __forceinline__ v16bf frag_a(const bf16* p, int ld, int lane) {
    int r = lane & 15, h = lane >> 4;
    const uint4* p0 = (const uint4*)(p + r * ld + h * 8);
    const uint4* p1 = (const uint4*)(p + r * ld + h * 8 + 16);
    union { uint4 q[2]; v16bf v; } u;
    u.q[0] = *p0; u.q[1] = *p1;
    return u.v;
}

// B-fragment 32x16 bf16 loaded from B^T stored row-major (row = output col,
// contiguous along K), leading dim ld. lane L: col = L&15, h=L>>4: elems 0..7 =
// K h*8..h*8+7, elems 8..15 = K 16+h*8 .. 16+h*8+7 (same as frag_a).
__device__ __forceinline__ v16bf frag_bT(const bf16* p, int ld, int lane) {
    int c = lane & 15, h = lane >> 4;
    const uint4* p0 = (const uint4*)(p + c * ld + h * 8);
    const uint4* p1 = (const uint4*)(p + c * ld + h * 8 + 16);
    union { uint4 q[2]; v16bf v; } u;
    u.q[0] = *p0; u.q[1] = *p1;
    return u.v;
}

__device__ __forceinline__ v8f wmma_bf16(v16bf a, v16bf b, v8f c) {
    return __builtin_amdgcn_wmma_f32_16x16x32_bf16(
        /*neg_a=*/false, a, /*neg_b=*/false, b,
        /*c_mod=*/(short)0, c, /*reuse_a=*/false, /*reuse_b=*/false);
}

__device__ __forceinline__ v8f vzero8() {
    v8f z;
#pragma unroll
    for (int j = 0; j < 8; ++j) z[j] = 0.0f;
    return z;
}

// ---- elementwise f32 -> bf16 cast -----------------------------------------
__global__ void cast_bf16_kernel(const float* __restrict__ in,
                                 bf16* __restrict__ out, int n) {
    int i = blockIdx.x * 256 + threadIdx.x;
    if (i < n) out[i] = (bf16)in[i];
}

// ---- per-node folded GCN weights: A, Bv -----------------------------------
// A[n,o] = sum_d emb[nt[n],d] * sum_i w_in[i]*wp[d,i,o]  (Bv with b_in)
__global__ void __launch_bounds__(256)
node_weight_kernel(const int* __restrict__ node_type,
                   const float* __restrict__ node_emb,
                   const float* __restrict__ wp,
                   const float* __restrict__ w_in,
                   const float* __restrict__ b_in,
                   float* __restrict__ A, float* __restrict__ Bv) {
    int tid = threadIdx.x;
    int nl = tid >> 6, o = tid & 63;
    int n = blockIdx.x * 4 + nl;
    int ty = node_type[n];
    float a = 0.f, bb = 0.f;
#pragma unroll
    for (int d = 0; d < 8; ++d) {
        float e = node_emb[ty * 8 + d];
        float sa = 0.f, sb = 0.f;
        for (int i = 0; i < 64; ++i) {
            float w = wp[(d * 64 + i) * 64 + o];
            sa += w_in[i] * w;
            sb += b_in[i] * w;
        }
        a += e * sa; bb += e * sb;
    }
    A[n * 64 + o] = a;
    Bv[n * 64 + o] = bb;
}

// ---- adj row sums ----------------------------------------------------------
__global__ void __launch_bounds__(256)
rowsum_kernel(const float* __restrict__ adj, float* __restrict__ r) {
    __shared__ float red[256];
    int n = blockIdx.x;
    float s = 0.f;
    for (int m = threadIdx.x; m < NN; m += 256) s += adj[(size_t)n * NN + m];
    red[threadIdx.x] = s;
    __syncthreads();
    for (int st = 128; st > 0; st >>= 1) {
        if ((int)threadIdx.x < st) red[threadIdx.x] += red[threadIdx.x + st];
        __syncthreads();
    }
    if (threadIdx.x == 0) r[n] = red[0];
}

// ---- S = adj @ X : (1024 x 768), K = 1024, bf16 WMMA ----------------------
// block = 256 thr = 8 waves; block tile 128(M) x 128(N); wave tile 32x64.
__global__ void __launch_bounds__(256)
gemm_s_kernel(const bf16* __restrict__ adjb, const bf16* __restrict__ xb,
              float* __restrict__ S) {
    int tid = threadIdx.x, lane = tid & 31, wave = tid >> 5;
    int wm = wave & 3, wn = wave >> 2;
    int m0 = blockIdx.x * 128 + wm * 32;   // node rows
    int n0 = blockIdx.y * 128 + wn * 64;   // bt cols

    v8f acc[2][4];
#pragma unroll
    for (int mi = 0; mi < 2; ++mi)
#pragma unroll
        for (int nt = 0; nt < 4; ++nt) acc[mi][nt] = vzero8();

    for (int k0 = 0; k0 < NN; k0 += 32) {
        v16bf a0 = frag_a(adjb + (size_t)m0 * NN + k0, NN, lane);
        v16bf a1 = frag_a(adjb + (size_t)(m0 + 16) * NN + k0, NN, lane);
#pragma unroll
        for (int nt = 0; nt < 4; ++nt) {
            v16bf b = frag_bT(xb + (size_t)(n0 + nt * 16) * NN + k0, NN, lane);
            acc[0][nt] = wmma_bf16(a0, b, acc[0][nt]);
            acc[1][nt] = wmma_bf16(a1, b, acc[1][nt]);
        }
    }
    int rb = (lane >> 4) * 8, col = lane & 15;
#pragma unroll
    for (int mi = 0; mi < 2; ++mi)
#pragma unroll
        for (int nt = 0; nt < 4; ++nt)
#pragma unroll
            for (int v = 0; v < 8; ++v)
                S[(size_t)(m0 + mi * 16 + rb + v) * BT + n0 + nt * 16 + col] =
                    acc[mi][nt][v];
}

// ---- fused GCN tail + GroupNorm, write hn[b,n,t,c] as bf16 ----------------
// one block per bt; thread owns fixed channel c = tid&63, strides over n.
__global__ void __launch_bounds__(256)
gcn_gn_kernel(const float* __restrict__ S, const float* __restrict__ A,
              const float* __restrict__ Bv, const float* __restrict__ r,
              const float* __restrict__ x, const float* __restrict__ w_in,
              const float* __restrict__ b_in, const float* __restrict__ gn_w,
              const float* __restrict__ gn_b, bf16* __restrict__ hn) {
    __shared__ float red[256], red2[256];
    __shared__ float mu_s[4], rs_s[4];
    int bt = blockIdx.x;
    int b = bt / TT, t = bt - b * TT;
    int tid = threadIdx.x;
    int c = tid & 63, g = c >> 4;
    float wv = w_in[c], biv = b_in[c], gw = gn_w[c], gb = gn_b[c];
    const float* xrow = x + (size_t)bt * NN;

    float sum = 0.f, sq = 0.f;
    for (int n = tid >> 6; n < NN; n += 4) {
        float sv = S[(size_t)n * BT + bt];
        float val = sv * A[n * 64 + c] + r[n] * Bv[n * 64 + c] + xrow[n] * wv + biv;
        sum += val; sq += val * val;
    }
    red[tid] = sum; red2[tid] = sq;
    __syncthreads();
    if (tid < 4) {
        float s0 = 0.f, s1 = 0.f;
        for (int q = 0; q < 4; ++q)
            for (int j = 0; j < 16; ++j) {
                int idx = q * 64 + tid * 16 + j;
                s0 += red[idx]; s1 += red2[idx];
            }
        float mu = s0 * (1.0f / 16384.0f);
        float var = s1 * (1.0f / 16384.0f) - mu * mu;
        mu_s[tid] = mu;
        rs_s[tid] = rsqrtf(var + 1e-5f);
    }
    __syncthreads();
    float mu = mu_s[g], rstd = rs_s[g];
    for (int n = tid >> 6; n < NN; n += 4) {
        float sv = S[(size_t)n * BT + bt];
        float val = sv * A[n * 64 + c] + r[n] * Bv[n * 64 + c] + xrow[n] * wv + biv;
        float o = (val - mu) * rstd * gw + gb;
        hn[((size_t)(b * NN + n) * TT + t) * CC + c] = (bf16)o;
    }
}

// ---- temporal mixer + SiLU + out projection, one block per (b,n) ----------
// 128 thr = 4 waves. GEMM1: Z=relu(Ht@W1^T+b1) (64x64, K=96);
// GEMM2: Tt=Z@W2^T+b2 (64x96, K=64). Wave w owns row tile 16w..16w+15.
__global__ void __launch_bounds__(128)
mixer_kernel(const bf16* __restrict__ hn, const bf16* __restrict__ w1b,
             const float* __restrict__ b1, const bf16* __restrict__ w2b,
             const float* __restrict__ b2, const float* __restrict__ w_out,
             const float* __restrict__ b_out, float* __restrict__ out) {
    __shared__ __align__(16) bf16 lds_ht[CC * TT];   // Ht[c][t]
    __shared__ __align__(16) bf16 lds_z[CC * HH];    // Z[c][h]
    __shared__ __align__(16) float lds_t[CC * TT];   // Tt[c][t]

    int bn = blockIdx.x;
    int b = bn >> 10, n = bn & (NN - 1);
    int tid = threadIdx.x, lane = tid & 31, w = tid >> 5;
    const bf16* tile = hn + (size_t)bn * (TT * CC);  // hn[t][c]

    // load hn tile (t,c) -> transposed Ht (c,t) in LDS, 8 bf16 per load
    for (int idx = tid; idx < TT * 8; idx += 128) {
        int t = idx >> 3, c8 = (idx & 7) << 3;
        union { uint4 q; bf16 e[8]; } u;
        u.q = *(const uint4*)(tile + t * CC + c8);
#pragma unroll
        for (int j = 0; j < 8; ++j) lds_ht[(c8 + j) * TT + t] = u.e[j];
    }
    __syncthreads();

    // GEMM1: Z[16w..16w+15, :] = relu(Ht @ W1^T + b1)
    v8f acc1[4];
#pragma unroll
    for (int nt = 0; nt < 4; ++nt) acc1[nt] = vzero8();
#pragma unroll
    for (int ks = 0; ks < 3; ++ks) {
        v16bf a = frag_a(lds_ht + (16 * w) * TT + ks * 32, TT, lane);
#pragma unroll
        for (int nt = 0; nt < 4; ++nt) {
            v16bf bb = frag_bT(w1b + (nt * 16) * TT + ks * 32, TT, lane);
            acc1[nt] = wmma_bf16(a, bb, acc1[nt]);
        }
    }
    int rb = (lane >> 4) * 8, col = lane & 15;
#pragma unroll
    for (int nt = 0; nt < 4; ++nt) {
        int h = nt * 16 + col;
        float bias = b1[h];
#pragma unroll
        for (int v = 0; v < 8; ++v) {
            float z = acc1[nt][v] + bias;
            z = z > 0.f ? z : 0.f;
            lds_z[(16 * w + rb + v) * HH + h] = (bf16)z;
        }
    }
    __syncthreads();

    // GEMM2: Tt[16w..16w+15, :] = Z @ W2^T + b2
    v8f acc2[6];
#pragma unroll
    for (int nt = 0; nt < 6; ++nt) acc2[nt] = vzero8();
#pragma unroll
    for (int ks = 0; ks < 2; ++ks) {
        v16bf a = frag_a(lds_z + (16 * w) * HH + ks * 32, HH, lane);
#pragma unroll
        for (int nt = 0; nt < 6; ++nt) {
            v16bf bb = frag_bT(w2b + (nt * 16) * HH + ks * 32, HH, lane);
            acc2[nt] = wmma_bf16(a, bb, acc2[nt]);
        }
    }
#pragma unroll
    for (int nt = 0; nt < 6; ++nt) {
        int tcol = nt * 16 + col;
        float bias = b2[tcol];
#pragma unroll
        for (int v = 0; v < 8; ++v)
            lds_t[(16 * w + rb + v) * TT + tcol] = acc2[nt][v] + bias;
    }
    __syncthreads();

    // y[t] = b_out + sum_c silu(Ht[c,t] + Tt[c,t]) * w_out[c]
    if (tid < TT) {
        int t = tid;
        float y = 0.f;
        for (int c = 0; c < CC; ++c) {
            float hv = (float)lds_ht[c * TT + t] + lds_t[c * TT + t];
            float sg = hv / (1.0f + expf(-hv));
            y += sg * w_out[c];
        }
        out[(size_t)(b * TT + t) * NN + n] = y + b_out[0];
    }
}

// ---------------------------------------------------------------------------
extern "C" void kernel_launch(void* const* d_in, const int* in_sizes, int n_in,
                              void* d_out, int out_size, void* d_ws, size_t ws_size,
                              hipStream_t stream) {
    const float* x        = (const float*)d_in[0];
    const float* adj      = (const float*)d_in[1];
    const int*   node_type= (const int*)  d_in[2];
    const float* w_in     = (const float*)d_in[3];
    const float* b_in     = (const float*)d_in[4];
    const float* wp       = (const float*)d_in[5];
    const float* node_emb = (const float*)d_in[6];
    const float* gn_w     = (const float*)d_in[7];
    const float* gn_b     = (const float*)d_in[8];
    const float* w1       = (const float*)d_in[9];
    const float* b1       = (const float*)d_in[10];
    const float* w2       = (const float*)d_in[11];
    const float* b2       = (const float*)d_in[12];
    const float* w_out    = (const float*)d_in[13];
    const float* b_out    = (const float*)d_in[14];
    float* out = (float*)d_out;

    char* ws = (char*)d_ws;
    size_t off = 0;
    bf16*  adjb = (bf16*)(ws + off);  off += (size_t)NN * NN * 2;        // 2 MB
    bf16*  xb   = (bf16*)(ws + off);  off += (size_t)BT * NN * 2;        // 1.5 MB
    bf16*  w1b  = (bf16*)(ws + off);  off += (size_t)HH * TT * 2;        // 12 KB
    bf16*  w2b  = (bf16*)(ws + off);  off += (size_t)TT * HH * 2;        // 12 KB
    float* S    = (float*)(ws + off); off += (size_t)NN * BT * 4;        // 3 MB
    float* A    = (float*)(ws + off); off += (size_t)NN * CC * 4;        // 256 KB
    float* Bv   = (float*)(ws + off); off += (size_t)NN * CC * 4;        // 256 KB
    float* r    = (float*)(ws + off); off += (size_t)NN * 4;             // 4 KB
    bf16*  hn   = (bf16*)(ws + off);  off += (size_t)BN * TT * CC * 2;   // 96 MB

    // casts to bf16
    cast_bf16_kernel<<<(NN * NN + 255) / 256, 256, 0, stream>>>(adj, adjb, NN * NN);
    cast_bf16_kernel<<<(BT * NN + 255) / 256, 256, 0, stream>>>(x, xb, BT * NN);
    cast_bf16_kernel<<<(HH * TT + 255) / 256, 256, 0, stream>>>(w1, w1b, HH * TT);
    cast_bf16_kernel<<<(TT * HH + 255) / 256, 256, 0, stream>>>(w2, w2b, TT * HH);

    // per-node folded weights + adj row sums
    node_weight_kernel<<<NN / 4, 256, 0, stream>>>(node_type, node_emb, wp,
                                                   w_in, b_in, A, Bv);
    rowsum_kernel<<<NN, 256, 0, stream>>>(adj, r);

    // S = adj @ X (WMMA)
    gemm_s_kernel<<<dim3(NN / 128, BT / 128), 256, 0, stream>>>(adjb, xb, S);

    // fused GCN tail + GroupNorm -> hn (bf16, temporal layout)
    gcn_gn_kernel<<<BT, 256, 0, stream>>>(S, A, Bv, r, x, w_in, b_in,
                                          gn_w, gn_b, hn);

    // fused temporal MLP (WMMA x2) + SiLU + out projection
    mixer_kernel<<<BN, 128, 0, stream>>>(hn, w1b, b1, w2b, b2, w_out, b_out, out);
}
